// QuantInternMLP_47571057771143
// MI455X (gfx1250) — 8-bit matrix operand order repaired and checked by reading, compiled, not run on hardware
//
#include <hip/hip_runtime.h>


typedef __attribute__((ext_vector_type(8))) int v8i;
typedef __attribute__((ext_vector_type(4))) int v4i;

typedef __attribute__((address_space(1))) v4i gas_v4i;
typedef __attribute__((address_space(3))) v4i las_v4i;

__device__ __forceinline__ void async_copy16(const void* g, void* l) {
#if __has_builtin(__builtin_amdgcn_global_load_async_to_lds_b128)
  __builtin_amdgcn_global_load_async_to_lds_b128(
      (gas_v4i*)(void*)g, (las_v4i*)l, 0, 0);
#else
  *(uint4*)l = *(const uint4*)g;
#endif
}

__device__ __forceinline__ void wait_async() {
#if __has_builtin(__builtin_amdgcn_s_wait_asynccnt)
  __builtin_amdgcn_s_wait_asynccnt(0);
#else
  asm volatile("s_wait_asynccnt 0x0" ::: "memory");
#endif
}

__device__ __forceinline__ float gelu_exact(float x) {
  return 0.5f * x * (1.0f + erff(x * 0.70710678118654752440f));
}

__global__ __launch_bounds__(256) void pack_i8_kernel(const int* __restrict__ src,
                                                      unsigned int* __restrict__ dst,
                                                      long long n4) {
  long long i = (long long)blockIdx.x * 256 + threadIdx.x;
  if (i >= n4) return;
  int4 v = ((const int4*)src)[i];
  unsigned int p = (v.x & 0xff) | ((v.y & 0xff) << 8) |
                   ((v.z & 0xff) << 16) | ((v.w & 0xff) << 24);
  dst[i] = p;
}

template <bool OUT_F16>
__global__ __launch_bounds__(256) void gemm_w8a8_kernel(
    const signed char* __restrict__ A8,
    const signed char* __restrict__ B8,
    const _Float16* __restrict__ scale_m,
    const float* __restrict__ scale_n,
    const _Float16* __restrict__ bias_n,
    void* __restrict__ out,
    int M_valid, int N, int K) {
  __shared__ signed char sA[2][128 * 128];
  __shared__ signed char sB[2][128 * 128];

  const int tid  = threadIdx.x;
  const int wave = tid >> 5;
  const int lane = tid & 31;
  const int lm   = lane & 15;
  const int lh   = lane >> 4;

  const int mBlock = blockIdx.y * 128;
  const int nBlock = blockIdx.x * 128;
  const int mw = (wave >> 1) * 32;
  const int nw = (wave & 1) * 64;

  const signed char* gA = A8 + (size_t)mBlock * K;
  const signed char* gB = B8 + (size_t)nBlock * K;

  auto stage_tiles = [&](int k0, int buf) {
#pragma unroll
    for (int it = 0; it < 4; ++it) {
      const int e   = tid + it * 256;
      const int row = e >> 3;
      const int c   = (e & 7) * 16;
      async_copy16(gA + (size_t)row * K + k0 + c, &sA[buf][row * 128 + c]);
      async_copy16(gB + (size_t)row * K + k0 + c, &sB[buf][row * 128 + c]);
    }
  };

  v8i acc[2][4] = {};

  stage_tiles(0, 0);

  for (int k0 = 0; k0 < K; k0 += 128) {
    const int cur = (k0 >> 7) & 1;
    wait_async();
    __syncthreads();

    if (k0 + 128 < K) stage_tiles(k0 + 128, cur ^ 1);

#pragma unroll
    for (int kk = 0; kk < 128; kk += 64) {

      v8i a[2];
#pragma unroll
      for (int i = 0; i < 2; ++i) {
        const signed char* p =
            &sA[cur][0] + (mw + i * 16 + lm) * 128 + kk + lh * 8;
        uint2 t0 = *(const uint2*)(p);
        uint2 t1 = *(const uint2*)(p + 16);
        uint2 t2 = *(const uint2*)(p + 32);
        uint2 t3 = *(const uint2*)(p + 48);
        a[i][0] = (int)t0.x; a[i][1] = (int)t0.y;
        a[i][2] = (int)t1.x; a[i][3] = (int)t1.y;
        a[i][4] = (int)t2.x; a[i][5] = (int)t2.y;
        a[i][6] = (int)t3.x; a[i][7] = (int)t3.y;
      }

      v8i b[4];
#pragma unroll
      for (int j = 0; j < 4; ++j) {
        const signed char* q =
            &sB[cur][0] + (nw + j * 16 + lm) * 128 + kk + lh * 8;
        uint2 s0 = *(const uint2*)(q);
        uint2 s1 = *(const uint2*)(q + 16);
        uint2 s2 = *(const uint2*)(q + 32);
        uint2 s3 = *(const uint2*)(q + 48);
        b[j][0] = (int)s0.x; b[j][1] = (int)s0.y;
        b[j][2] = (int)s1.x; b[j][3] = (int)s1.y;
        b[j][4] = (int)s2.x; b[j][5] = (int)s2.y;
        b[j][6] = (int)s3.x; b[j][7] = (int)s3.y;
      }

#pragma unroll
      for (int j = 0; j < 4; ++j)
#pragma unroll
        for (int i = 0; i < 2; ++i)
          acc[i][j] = __builtin_amdgcn_wmma_i32_16x16x64_iu8(
               true, a[i],  true, b[j], acc[i][j],
               false,  false);
    }
  }

#pragma unroll
  for (int i = 0; i < 2; ++i) {
    const int rowBase = mBlock + mw + i * 16 + lh * 8;
#pragma unroll
    for (int j = 0; j < 4; ++j) {
      const int col = nBlock + nw + j * 16 + lm;
      const float sn = scale_n[col];
      const float bn = (float)bias_n[col];
#pragma unroll
      for (int r = 0; r < 8; ++r) {
        const int row = rowBase + r;
        if (row < M_valid) {
          const float sm = (float)scale_m[row];
          const float v = (float)acc[i][j][r] * sm * sn + bn;
          if constexpr (OUT_F16)
            ((_Float16*)out)[(size_t)row * N + col] = (_Float16)v;
          else
            ((float*)out)[(size_t)row * N + col] = v;
        }
      }
    }
  }
}

__global__ __launch_bounds__(256) void gelu_quant_kernel(
    const _Float16* __restrict__ fc1,
    unsigned int* __restrict__ act8,
    _Float16* __restrict__ scale_act,
    int I) {
  const int m = blockIdx.x;
  const int tid = threadIdx.x;
  const _Float16* row = fc1 + (size_t)m * I;

  __shared__ float red[256];

  float amax = 0.0f;
  for (int idx = tid; idx < I; idx += 256) {
    float g = gelu_exact((float)row[idx]);
    amax = fmaxf(amax, fabsf(g));
  }
  red[tid] = amax;
  __syncthreads();
#pragma unroll
  for (int s = 128; s > 0; s >>= 1) {
    if (tid < s) red[tid] = fmaxf(red[tid], red[tid + s]);
    __syncthreads();
  }
  const _Float16 sh = (_Float16)(red[0] * (1.0f / 127.0f));
  const float s = fmaxf((float)sh, 1e-8f);
  const float inv = 1.0f / s;
  if (tid == 0) scale_act[m] = sh;

  unsigned int* orow = act8 + ((size_t)m * I >> 2);
  for (int base = tid * 4; base < I; base += 1024) {
    unsigned int p = 0;
#pragma unroll
    for (int t = 0; t < 4; ++t) {
      float g = gelu_exact((float)row[base + t]);
      int q = (int)__builtin_rintf(g * inv);
      q = q > 127 ? 127 : (q < -128 ? -128 : q);
      p |= ((unsigned int)(q & 0xff)) << (8 * t);
    }
    orow[base >> 2] = p;
  }
}

extern "C" void kernel_launch(void* const* d_in, const int* in_sizes, int n_in,
                              void* d_out, int out_size, void* d_ws, size_t ws_size,
                              hipStream_t stream) {
  (void)in_sizes; (void)n_in; (void)out_size; (void)ws_size;

  const int*      hs       = (const int*)d_in[0];
  const _Float16* scale_in = (const _Float16*)d_in[1];
  const int*      w1       = (const int*)d_in[2];
  const float*    w1_scale = (const float*)d_in[3];
  const _Float16* b1       = (const _Float16*)d_in[4];
  const int*      w2       = (const int*)d_in[5];
  const float*    w2_scale = (const float*)d_in[6];
  const _Float16* b2       = (const _Float16*)d_in[7];
  float*          out      = (float*)d_out;

  constexpr int H  = 3200;
  constexpr int I  = 12800;
  constexpr int T  = 2 * 1025;
  constexpr int Tp = 2176;

  char* ws = (char*)d_ws;
  size_t off = 0;
  auto carve = [&](size_t bytes) {
    char* p = ws + off;
    off = (off + bytes + 255) & ~(size_t)255;
    return p;
  };
  signed char* x8    = (signed char*)carve((size_t)Tp * H);
  signed char* w1_8  = (signed char*)carve((size_t)I * H);
  signed char* w2_8  = (signed char*)carve((size_t)H * I);
  _Float16*    fc1   = (_Float16*)  carve((size_t)T * I * 2);
  signed char* act8  = (signed char*)carve((size_t)Tp * I);
  _Float16*    sact  = (_Float16*)  carve((size_t)Tp * 2);

  {
    long long n4 = (long long)T * H / 4;
    pack_i8_kernel<<<(unsigned)((n4 + 255) / 256), 256, 0, stream>>>(
        hs, (unsigned int*)x8, n4);
  }
  {
    long long n4 = (long long)I * H / 4;
    pack_i8_kernel<<<(unsigned)((n4 + 255) / 256), 256, 0, stream>>>(
        w1, (unsigned int*)w1_8, n4);
    pack_i8_kernel<<<(unsigned)((n4 + 255) / 256), 256, 0, stream>>>(
        w2, (unsigned int*)w2_8, n4);
  }

  gemm_w8a8_kernel<true><<<dim3(I / 128, Tp / 128), 256, 0, stream>>>(
      x8, w1_8, scale_in, w1_scale, b1, (void*)fc1, T, I, H);

  gelu_quant_kernel<<<T, 256, 0, stream>>>(fc1, (unsigned int*)act8, sact, I);

  gemm_w8a8_kernel<false><<<dim3(H / 128, Tp / 128), 256, 0, stream>>>(
      act8, w2_8, sact, w2_scale, b2, (void*)out, T, H, I);
}
